// SORFLayer_56753697849678
// MI455X (gfx1250) — matrix operand order repaired and checked by reading, not compiled, not run on hardware
//
#include <hip/hip_runtime.h>

// ---------------------------------------------------------------------------
// SORF layer for MI455X (gfx1250):
//   C[16384,4096] = Apad[16384,2048] x S[4096,2048]^T  (bf16 WMMA, f32 accum)
//   out[:, 0:4096]  = cos(C)/64
//   out[:, 4096:8192] = sin(C)/64
// ---------------------------------------------------------------------------

typedef __attribute__((ext_vector_type(16))) __bf16 v16bf;
typedef __attribute__((ext_vector_type(8)))  float  v8f;
typedef __attribute__((ext_vector_type(4)))  unsigned int v4u;

#define BATCH   16384
#define KDIM    2048
#define NDIM    4096      // 2 blocks * 2048
#define XCOLS   1280
#define BM      128
#define BN      128
#define BK      32
#define LDT     40        // LDS row stride in bf16 (80B) -> conflict-free b128 reads

__device__ __forceinline__ v16bf pack16(v4u a, v4u b) {
    struct P { v4u lo, hi; } p{a, b};
    return __builtin_bit_cast(v16bf, p);
}

// ---- pass 1: x (fp32 [16384,1280]) -> bf16 [16384,2048], zero padded -------
__global__ __launch_bounds__(256)
void sorf_cvt_x(const float* __restrict__ x, __bf16* __restrict__ A) {
    int i = (blockIdx.x * 256 + threadIdx.x) * 2;   // pair of elements
    if (i >= BATCH * KDIM) return;
    int row = i >> 11;          // /2048
    int col = i & (KDIM - 1);
    float v0 = (col + 0 < XCOLS) ? x[row * XCOLS + col + 0] : 0.0f;
    float v1 = (col + 1 < XCOLS) ? x[row * XCOLS + col + 1] : 0.0f;
    A[i + 0] = (__bf16)v0;
    A[i + 1] = (__bf16)v1;
}

// ---- pass 2: sorf (fp32 [2,2048,2048] == [4096,2048]) -> bf16 --------------
__global__ __launch_bounds__(256)
void sorf_cvt_s(const float* __restrict__ s, __bf16* __restrict__ B) {
    int i = (blockIdx.x * 256 + threadIdx.x) * 2;
    if (i >= NDIM * KDIM) return;
    B[i + 0] = (__bf16)s[i + 0];
    B[i + 1] = (__bf16)s[i + 1];
}

// ---- pass 3: tiled bf16 WMMA GEMM + fused cos/sin epilogue -----------------
__global__ __launch_bounds__(256)
void sorf_gemm(const __bf16* __restrict__ A,   // [BATCH, KDIM]
               const __bf16* __restrict__ B,   // [NDIM,  KDIM]
               float* __restrict__ out) {      // [BATCH, 2*NDIM]
    __shared__ __bf16 lds_a[BM * LDT];
    __shared__ __bf16 lds_b[BN * LDT];

    const int t    = threadIdx.x;
    const int lane = t & 31;
    const int wave = t >> 5;          // 0..7
    const int wm   = wave >> 1;       // 0..3  -> 32-row strip
    const int wn   = wave & 1;        // 0..1  -> 64-col strip
    const int hi   = lane >> 4;       // half-wave select
    const int l16  = lane & 15;

    const int m0 = blockIdx.x * BM;
    const int n0 = blockIdx.y * BN;

    // staging assignment: thread t loads 16 contiguous bf16 (32B) of row t/2
    const int ldRow = t >> 1;             // 0..127
    const int ldCol = (t & 1) << 4;       // 0 or 16
    const __bf16* gA = A + (size_t)(m0 + ldRow) * KDIM + ldCol;
    const __bf16* gB = B + (size_t)(n0 + ldRow) * KDIM + ldCol;

    v8f acc[8] = {};                      // 2 (M) x 4 (N) tiles of 16x16

    // prefetch k-tile 0
    v4u ra0 = *(const v4u*)(gA + 0);
    v4u ra1 = *(const v4u*)(gA + 8);
    v4u rb0 = *(const v4u*)(gB + 0);
    v4u rb1 = *(const v4u*)(gB + 8);

    for (int kt = 0; kt < KDIM; kt += BK) {
        // regs -> LDS
        *(v4u*)&lds_a[ldRow * LDT + ldCol + 0] = ra0;
        *(v4u*)&lds_a[ldRow * LDT + ldCol + 8] = ra1;
        *(v4u*)&lds_b[ldRow * LDT + ldCol + 0] = rb0;
        *(v4u*)&lds_b[ldRow * LDT + ldCol + 8] = rb1;
        __syncthreads();

        // prefetch next k-tile (overlaps with WMMA below)
        if (kt + BK < KDIM) {
            ra0 = *(const v4u*)(gA + kt + BK + 0);
            ra1 = *(const v4u*)(gA + kt + BK + 8);
            rb0 = *(const v4u*)(gB + kt + BK + 0);
            rb1 = *(const v4u*)(gB + kt + BK + 8);
        }

        // A fragments: lane holds K {h*8..h*8+7, h*8+16..h*8+23}, h = lane>=16
        v16bf af[2];
#pragma unroll
        for (int i = 0; i < 2; ++i) {
            const __bf16* pa = &lds_a[(wm * 32 + i * 16 + l16) * LDT + hi * 8];
            af[i] = pack16(*(const v4u*)pa, *(const v4u*)(pa + 16));
        }
        // B fragments: lane holds K {h*8..h*8+7, h*8+16..h*8+23}, h = lane>=16
        v16bf bf[4];
#pragma unroll
        for (int j = 0; j < 4; ++j) {
            const __bf16* pb = &lds_b[(wn * 64 + j * 16 + l16) * LDT + hi * 8];
            bf[j] = pack16(*(const v4u*)pb, *(const v4u*)(pb + 16));
        }
#pragma unroll
        for (int i = 0; i < 2; ++i)
#pragma unroll
            for (int j = 0; j < 4; ++j)
                acc[i * 4 + j] = __builtin_amdgcn_wmma_f32_16x16x32_bf16(
                    false, af[i], false, bf[j], (short)0, acc[i * 4 + j],
                    false, false);

        __syncthreads();
    }

    // fused epilogue: out = {cos,sin}(z)/sqrt(4096); v_cos/v_sin take revs
    const float INV2PI = 0.15915494309189535f;
    const float SCALE  = 0.015625f;   // 1/64
#pragma unroll
    for (int i = 0; i < 2; ++i) {
        const int rbase = m0 + wm * 32 + i * 16 + hi * 8;
#pragma unroll
        for (int j = 0; j < 4; ++j) {
            const int col = n0 + wn * 64 + j * 16 + l16;
            v8f c = acc[i * 4 + j];
#pragma unroll
            for (int e = 0; e < 8; ++e) {
                float zr = c[e] * INV2PI;
                float cs = __builtin_amdgcn_cosf(zr) * SCALE;
                float sn = __builtin_amdgcn_sinf(zr) * SCALE;
                size_t o = (size_t)(rbase + e) * (2 * NDIM) + col;
                out[o]        = cs;
                out[o + NDIM] = sn;
            }
        }
    }
}

extern "C" void kernel_launch(void* const* d_in, const int* in_sizes, int n_in,
                              void* d_out, int out_size, void* d_ws, size_t ws_size,
                              hipStream_t stream) {
    const float* x    = (const float*)d_in[0];   // [16384,1280]
    const float* sorf = (const float*)d_in[1];   // [2,2048,2048]
    float* out        = (float*)d_out;           // [16384,8192]

    __bf16* Abf = (__bf16*)d_ws;                                   // 64 MB
    __bf16* Bbf = (__bf16*)((char*)d_ws + (size_t)BATCH * KDIM * 2); // +16 MB

    {   // x -> bf16 (padded)
        int pairs = BATCH * KDIM / 2;
        sorf_cvt_x<<<(pairs + 255) / 256, 256, 0, stream>>>(x, Abf);
    }
    {   // sorf -> bf16
        int pairs = NDIM * KDIM / 2;
        sorf_cvt_s<<<(pairs + 255) / 256, 256, 0, stream>>>(sorf, Bbf);
    }
    {
        dim3 grid(BATCH / BM, NDIM / BN);   // 128 x 32
        sorf_gemm<<<grid, 256, 0, stream>>>(Abf, Bbf, out);
    }
}
